// MyBert_42176578847313
// MI455X (gfx1250) — compile-verified, matrix operand order checked by reading, not run on hardware
//
#include <hip/hip_runtime.h>
#include <hip/hip_bf16.h>

// ---------------------------------------------------------------------------
// BERT-base forward on gfx1250 (MI455X). All GEMMs via v_wmma_f32_16x16x32_f16
// (fp32 -> f16 conversion while staging into LDS; fp32 accumulate).
// GEMM: 128x128 block tile, 32x64 wave tile, double-buffered LDS, b128
// register staging with scheduling fences, global_prefetch of tile k+2.
// ---------------------------------------------------------------------------

#define B_   32
#define S_   128
#define H_   768
#define NH_  12
#define DH_  64
#define FF_  3072
#define L_   12
#define NTOK (B_ * S_)     // 4096
#define LW_  64
#define NI_  26
#define NS_  130

typedef _Float16 v16h __attribute__((ext_vector_type(16)));
typedef _Float16 v8h  __attribute__((ext_vector_type(8)));
typedef _Float16 v4h  __attribute__((ext_vector_type(4)));
typedef float    v8f  __attribute__((ext_vector_type(8)));
typedef float    v4f  __attribute__((ext_vector_type(4)));

union Frag { v16h f; v8h h[2]; };

__device__ __forceinline__ v8f wmma_f16(const Frag& a, const Frag& b, v8f c) {
    return __builtin_amdgcn_wmma_f32_16x16x32_f16(
        /*neg_a=*/false, a.f, /*neg_b=*/false, b.f,
        /*c_mod=*/(short)0, c, /*reuse_a=*/false, /*reuse_b=*/false);
}

#define SCHED_FENCE() asm volatile("" ::: "memory")

// ---------------------------------------------------------------------------
// Generic GEMM: C[M,N] = A[M,K] * W[K,N] + bias[N]   (optional exact GELU)
// block = 256 threads (8 waves, 4x2), block tile 128x128, wave tile 32x64,
// K-step 32. Double-buffered LDS; loads for step k+1 overlap WMMAs on step k.
// ---------------------------------------------------------------------------
#define BM 128
#define BN 128
#define BK 32
#define LDS_A 40   // BK + 8 halves pad (80 B rows, 16B-aligned)
#define LDS_B 40

__global__ __launch_bounds__(256) void gemm_f16w_kernel(
    const float* __restrict__ A, const float* __restrict__ W,
    const float* __restrict__ bias, float* __restrict__ C,
    int M, int N, int K, int gelu)
{
    __shared__ __align__(16) _Float16 As[2][BM * LDS_A];   // [m][k]
    __shared__ __align__(16) _Float16 Bs[2][BN * LDS_B];   // [n][k] (transposed)

    const int tid  = threadIdx.x;
    const int lane = tid & 31;
    const int wave = tid >> 5;     // 0..7
    const int wm   = wave >> 1;    // 0..3  (32-row slab)
    const int wn   = wave & 1;     // 0..1  (64-col slab)
    const int row0 = blockIdx.y * BM;
    const int col0 = blockIdx.x * BN;

    const int fm = lane & 15;            // row(A) / col(B) inside 16-tile
    const int kg = (lane >> 4) << 3;     // 0 or 8: K-half select per ISA layout

    v4f ra[4];   // register-staged A tile (128x32: 4 x float4 per thread)
    v4f rb[4];   // register-staged W tile ( 32x128: 4 x float4 per thread)

    // issue wide global loads for the k0 tile into registers (b128 each)
    auto load_tile = [&](int k0) {
#pragma unroll
        for (int i = 0; i < 4; ++i) {
            int e = tid + i * 256;                    // 1024 float4s, row=e>>3
            ra[i] = *(const v4f*)&A[(size_t)(row0 + (e >> 3)) * K + k0 + (e & 7) * 4];
        }
#pragma unroll
        for (int i = 0; i < 4; ++i) {
            int e = tid + i * 256;                    // 1024 float4s, kk=e>>5
            rb[i] = *(const v4f*)&W[(size_t)(k0 + (e >> 5)) * N + col0 + (e & 31) * 4];
        }
    };

    // prefetch the tile at k0 into GL2 (weights streamed once -> pure win)
    auto prefetch_tile = [&](int k0) {
        if (k0 < K) {
            if (tid < BM)
                __builtin_prefetch(&A[(size_t)(row0 + tid) * K + k0], 0, 0);
            __builtin_prefetch(&W[(size_t)(k0 + (tid >> 3)) * N + col0 + (tid & 7) * 16], 0, 0);
        }
    };

    // convert + store register tile into LDS buffer `buf`
    auto store_tile = [&](int buf) {
#pragma unroll
        for (int i = 0; i < 4; ++i) {
            int e = tid + i * 256;
            int m = e >> 3, c = (e & 7) * 4;
            v4h p;
#pragma unroll
            for (int j = 0; j < 4; ++j) p[j] = (_Float16)ra[i][j];
            *(v4h*)&As[buf][m * LDS_A + c] = p;       // aligned ds_store_b64
        }
#pragma unroll
        for (int i = 0; i < 4; ++i) {
            int e = tid + i * 256;
            int kk = e >> 5, c = (e & 31) * 4;
#pragma unroll
            for (int j = 0; j < 4; ++j)               // transpose scatter
                Bs[buf][(c + j) * LDS_B + kk] = (_Float16)rb[i][j];
        }
    };

    v8f acc[2][4];
#pragma unroll
    for (int i = 0; i < 2; ++i)
#pragma unroll
        for (int j = 0; j < 4; ++j) acc[i][j] = (v8f)0.0f;

    auto compute = [&](int buf) {
#pragma unroll
        for (int tm = 0; tm < 2; ++tm) {
            Frag a;
            const _Float16* ap = &As[buf][(wm * 32 + tm * 16 + fm) * LDS_A];
            a.h[0] = *(const v8h*)(ap + kg);
            a.h[1] = *(const v8h*)(ap + 16 + kg);
#pragma unroll
            for (int tn = 0; tn < 4; ++tn) {
                Frag b;
                const _Float16* bp = &Bs[buf][(wn * 64 + tn * 16 + fm) * LDS_B];
                b.h[0] = *(const v8h*)(bp + kg);
                b.h[1] = *(const v8h*)(bp + 16 + kg);
                acc[tm][tn] = wmma_f16(a, b, acc[tm][tn]);
            }
        }
    };

    // --- software pipeline ---
    load_tile(0);
    SCHED_FENCE();
    store_tile(0);
    __syncthreads();

    int cur = 0;
    for (int k0 = BK; k0 < K; k0 += BK) {
        prefetch_tile(k0 + BK);     // pull tile k+2 toward GL2
        load_tile(k0);              // global b128 loads for tile k+1, in flight
        SCHED_FENCE();              // keep loads issued BEFORE the WMMA block
        compute(cur);               // WMMAs on tile k overlap the loads
        SCHED_FENCE();              // keep LDS fill AFTER the WMMA block
        store_tile(cur ^ 1);        // single loadcnt wait, cvt f16, fill buffer
        __syncthreads();
        cur ^= 1;
    }
    compute(cur);                   // drain last tile

    // epilogue: C layout = col = lane%16, row = v + 8*(lane/16)
    const int mh = (lane >> 4) << 3;
#pragma unroll
    for (int tm = 0; tm < 2; ++tm)
#pragma unroll
        for (int tn = 0; tn < 4; ++tn) {
            int gcol = col0 + wn * 64 + tn * 16 + fm;
            float bv = bias[gcol];
#pragma unroll
            for (int v = 0; v < 8; ++v) {
                int grow = row0 + wm * 32 + tm * 16 + mh + v;
                float x = acc[tm][tn][v] + bv;
                if (gelu) x = 0.5f * x * (1.0f + erff(x * 0.70710678118654752f));
                C[(size_t)grow * N + gcol] = x;
            }
        }
}

// ---------------------------------------------------------------------------
// Attention: one block per (b,h). scores = Q K^T /8 + mask; softmax; ctx = P V
// 8 waves, each owns 16 query rows. Softmax fully in registers (shfl_xor).
// ---------------------------------------------------------------------------
struct SMemQK { _Float16 q[S_ * 72]; _Float16 k[S_ * 72]; };   // [s][dh], pad 72
union  SMemU  { SMemQK qk; _Float16 p[S_ * 136]; };            // P: [q][key], pad 136

__global__ __launch_bounds__(256) void attention_kernel(
    const float* __restrict__ Q, const float* __restrict__ K,
    const float* __restrict__ V, const int* __restrict__ amask,
    float* __restrict__ Ctx)
{
    __shared__ __align__(16) SMemU u;
    __shared__ __align__(16) _Float16 vt[DH_ * 136];   // V transposed [dh][key]
    __shared__ float maskadd[S_];

    const int bh   = blockIdx.x;
    const int b    = bh / NH_;
    const int h    = bh % NH_;
    const int tid  = threadIdx.x;
    const int lane = tid & 31;
    const int wave = tid >> 5;

    // stage Q,K,V (f32 [token][768], head slice h*64..h*64+63) -> f16 LDS
    for (int i = tid; i < S_ * DH_; i += 256) {
        int s = i >> 6, d = i & 63;
        size_t g = (size_t)(b * S_ + s) * H_ + h * DH_ + d;
        u.qk.q[s * 72 + d] = (_Float16)Q[g];
        u.qk.k[s * 72 + d] = (_Float16)K[g];
        vt[d * 136 + s]    = (_Float16)V[g];
    }
    if (tid < S_)
        maskadd[tid] = (1.0f - (float)amask[b * S_ + tid]) * -10000.0f;
    __syncthreads();

    const int fm = lane & 15;
    const int kg = (lane >> 4) << 3;
    const int mh = (lane >> 4) << 3;

    // ---- scores: wave owns rows [wave*16, wave*16+16) x all 128 keys ----
    v8f acc[8];
#pragma unroll
    for (int t = 0; t < 8; ++t) acc[t] = (v8f)0.0f;

#pragma unroll
    for (int kk = 0; kk < DH_; kk += 32) {
        Frag a;
        const _Float16* ap = &u.qk.q[(wave * 16 + fm) * 72 + kk];
        a.h[0] = *(const v8h*)(ap + kg);
        a.h[1] = *(const v8h*)(ap + 16 + kg);
#pragma unroll
        for (int tn = 0; tn < 8; ++tn) {
            Frag bb;
            const _Float16* bp = &u.qk.k[(tn * 16 + fm) * 72 + kk];
            bb.h[0] = *(const v8h*)(bp + kg);
            bb.h[1] = *(const v8h*)(bp + 16 + kg);
            acc[tn] = wmma_f16(a, bb, acc[tn]);
        }
    }

    // ---- scale + mask, register softmax over keys ----
    float sc[8][8];
#pragma unroll
    for (int tn = 0; tn < 8; ++tn) {
        float ma = maskadd[tn * 16 + fm];
#pragma unroll
        for (int v = 0; v < 8; ++v)
            sc[tn][v] = acc[tn][v] * 0.125f + ma;
    }
    // row r = wave*16 + mh + v lives in 16 lanes of one half-group x 8 tiles
#pragma unroll
    for (int v = 0; v < 8; ++v) {
        float mx = sc[0][v];
#pragma unroll
        for (int tn = 1; tn < 8; ++tn) mx = fmaxf(mx, sc[tn][v]);
#pragma unroll
        for (int d = 8; d > 0; d >>= 1) mx = fmaxf(mx, __shfl_xor(mx, d, 32));
        float sum = 0.0f;
#pragma unroll
        for (int tn = 0; tn < 8; ++tn) {
            float e = __expf(sc[tn][v] - mx);
            sc[tn][v] = e;
            sum += e;
        }
#pragma unroll
        for (int d = 8; d > 0; d >>= 1) sum += __shfl_xor(sum, d, 32);
        float inv = 1.0f / sum;
#pragma unroll
        for (int tn = 0; tn < 8; ++tn) sc[tn][v] *= inv;
    }

    __syncthreads();   // Q/K fragments consumed -> safe to overwrite with P
#pragma unroll
    for (int tn = 0; tn < 8; ++tn)
#pragma unroll
        for (int v = 0; v < 8; ++v)
            u.p[(wave * 16 + mh + v) * 136 + tn * 16 + fm] = (_Float16)sc[tn][v];
    __syncthreads();

    // ---- ctx = P[128,128] * V[128,64] ----
    v8f acc2[4];
#pragma unroll
    for (int t = 0; t < 4; ++t) acc2[t] = (v8f)0.0f;

#pragma unroll
    for (int kk = 0; kk < S_; kk += 32) {
        Frag a;
        const _Float16* ap = &u.p[(wave * 16 + fm) * 136 + kk];
        a.h[0] = *(const v8h*)(ap + kg);
        a.h[1] = *(const v8h*)(ap + 16 + kg);
#pragma unroll
        for (int tn = 0; tn < 4; ++tn) {
            Frag bb;
            const _Float16* bp = &vt[(tn * 16 + fm) * 136 + kk];
            bb.h[0] = *(const v8h*)(bp + kg);
            bb.h[1] = *(const v8h*)(bp + 16 + kg);
            acc2[tn] = wmma_f16(a, bb, acc2[tn]);
        }
    }
#pragma unroll
    for (int tn = 0; tn < 4; ++tn)
#pragma unroll
        for (int v = 0; v < 8; ++v)
            Ctx[(size_t)(b * S_ + wave * 16 + mh + v) * H_ + h * DH_ + tn * 16 + fm]
                = acc2[tn][v];
}

// ---------------------------------------------------------------------------
// LayerNorm kernels (block per token, 256 threads)
// ---------------------------------------------------------------------------
__device__ __forceinline__ void block_reduce2(float& s, float& q, float* red) {
#pragma unroll
    for (int d = 16; d > 0; d >>= 1) {
        s += __shfl_xor(s, d, 32);
        q += __shfl_xor(q, d, 32);
    }
    int lane = threadIdx.x & 31, wave = threadIdx.x >> 5;
    if (lane == 0) { red[wave] = s; red[8 + wave] = q; }
    __syncthreads();
    if (threadIdx.x == 0) {
        float S = 0.0f, Q = 0.0f;
#pragma unroll
        for (int w = 0; w < 8; ++w) { S += red[w]; Q += red[8 + w]; }
        red[0] = S; red[8] = Q;
    }
    __syncthreads();
    s = red[0]; q = red[8];
}

__global__ __launch_bounds__(256) void embed_ln_kernel(
    const int* __restrict__ inp, const float* __restrict__ wemb,
    const float* __restrict__ pemb, const float* __restrict__ temb,
    const float* __restrict__ g, const float* __restrict__ be,
    float* __restrict__ out)
{
    int token = blockIdx.x;
    int s = token & (S_ - 1);
    int wid = inp[token];
    __shared__ float buf[H_];
    __shared__ float red[16];
    float ls = 0.0f, lq = 0.0f;
    for (int i = threadIdx.x; i < H_; i += 256) {
        float v = wemb[(size_t)wid * H_ + i] + pemb[(size_t)s * H_ + i] + temb[i];
        buf[i] = v; ls += v; lq += v * v;
    }
    __syncthreads();
    block_reduce2(ls, lq, red);
    float mean = ls * (1.0f / H_);
    float var  = lq * (1.0f / H_) - mean * mean;
    float rstd = rsqrtf(var + 1e-12f);
    for (int i = threadIdx.x; i < H_; i += 256)
        out[(size_t)token * H_ + i] = (buf[i] - mean) * rstd * g[i] + be[i];
}

__global__ __launch_bounds__(256) void add_ln_kernel(
    const float* __restrict__ resid, const float* __restrict__ y,
    const float* __restrict__ g, const float* __restrict__ be,
    float* __restrict__ out)
{
    int token = blockIdx.x;
    __shared__ float buf[H_];
    __shared__ float red[16];
    float ls = 0.0f, lq = 0.0f;
    for (int i = threadIdx.x; i < H_; i += 256) {
        float v = resid[(size_t)token * H_ + i] + y[(size_t)token * H_ + i];
        buf[i] = v; ls += v; lq += v * v;
    }
    __syncthreads();
    block_reduce2(ls, lq, red);
    float mean = ls * (1.0f / H_);
    float var  = lq * (1.0f / H_) - mean * mean;
    float rstd = rsqrtf(var + 1e-12f);
    for (int i = threadIdx.x; i < H_; i += 256)
        out[(size_t)token * H_ + i] = (buf[i] - mean) * rstd * g[i] + be[i];
}

// ---------------------------------------------------------------------------
// Heads (tiny: scalar dot products)
// ---------------------------------------------------------------------------
__global__ void intent_kernel(const float* __restrict__ x,
                              const float* __restrict__ Wi,
                              const float* __restrict__ bi,
                              float* __restrict__ out)
{
    int idx = blockIdx.x * blockDim.x + threadIdx.x;
    if (idx >= B_ * NI_) return;
    int b = idx / NI_, i = idx % NI_;
    const float* xr = x + (size_t)(b * S_) * H_;   // token 0 of batch b
    float acc = bi[i];
    for (int d = 0; d < H_; ++d) acc += xr[d] * Wi[(size_t)d * NI_ + i];
    out[idx] = acc;
}

__global__ void slot_kernel(const float* __restrict__ x,
                            const int* __restrict__ mapping,
                            const float* __restrict__ Ws,
                            const float* __restrict__ bs,
                            float* __restrict__ out)
{
    int idx = blockIdx.x * blockDim.x + threadIdx.x;
    if (idx >= B_ * NS_ * LW_) return;
    int b  = idx / (NS_ * LW_);
    int r  = idx % (NS_ * LW_);
    int sl = r / LW_;
    int l  = r % LW_;
    int tok = mapping[b * LW_ + l];
    const float* xr = x + (size_t)(b * S_ + tok) * H_;
    float acc = bs[sl];
    for (int d = 0; d < H_; ++d) acc += xr[d] * Ws[(size_t)d * NS_ + sl];
    out[B_ * NI_ + idx] = acc;   // slot_logits laid out (B, N_SLOT, LW)
}

// ---------------------------------------------------------------------------
// Host orchestration
// ---------------------------------------------------------------------------
extern "C" void kernel_launch(void* const* d_in, const int* in_sizes, int n_in,
                              void* d_out, int out_size, void* d_ws, size_t ws_size,
                              hipStream_t stream)
{
    (void)in_sizes; (void)n_in; (void)out_size; (void)ws_size;

    const int*   input   = (const int*)  d_in[0];
    const int*   amask   = (const int*)  d_in[1];
    const int*   mapping = (const int*)  d_in[2];
    const float* wemb    = (const float*)d_in[3];
    const float* pemb    = (const float*)d_in[4];
    const float* temb    = (const float*)d_in[5];   // use row 0
    const float* eln_s   = (const float*)d_in[6];
    const float* eln_b   = (const float*)d_in[7];
    const float* Wq      = (const float*)d_in[8];
    const float* bq      = (const float*)d_in[9];
    const float* Wk      = (const float*)d_in[10];
    const float* bk      = (const float*)d_in[11];
    const float* Wv      = (const float*)d_in[12];
    const float* bv      = (const float*)d_in[13];
    const float* Wo      = (const float*)d_in[14];
    const float* bo      = (const float*)d_in[15];
    const float* ln1s    = (const float*)d_in[16];
    const float* ln1b    = (const float*)d_in[17];
    const float* W1      = (const float*)d_in[18];
    const float* b1      = (const float*)d_in[19];
    const float* W2      = (const float*)d_in[20];
    const float* b2      = (const float*)d_in[21];
    const float* ln2s    = (const float*)d_in[22];
    const float* ln2b    = (const float*)d_in[23];
    const float* Wi      = (const float*)d_in[24];
    const float* bi      = (const float*)d_in[25];
    const float* Ws      = (const float*)d_in[26];
    const float* bs      = (const float*)d_in[27];

    float* out = (float*)d_out;
    float* ws  = (float*)d_ws;

    const size_t T = (size_t)NTOK * H_;     // 3,145,728 floats
    float* x   = ws;
    float* qb  = x   + T;
    float* kb  = qb  + T;
    float* vb  = kb  + T;
    float* ctx = vb  + T;
    float* yb  = ctx + T;
    float* ffb = yb  + T;                   // NTOK * FF_ floats

    dim3 blk(256);
    dim3 gH(H_ / BN, NTOK / BM);            // (6, 32)
    dim3 gF(FF_ / BN, NTOK / BM);           // (24, 32)

    embed_ln_kernel<<<NTOK, blk, 0, stream>>>(input, wemb, pemb, temb, eln_s, eln_b, x);

    for (int l = 0; l < L_; ++l) {
        const float* wq = Wq + (size_t)l * H_ * H_;
        const float* wk = Wk + (size_t)l * H_ * H_;
        const float* wv = Wv + (size_t)l * H_ * H_;
        const float* wo = Wo + (size_t)l * H_ * H_;
        const float* w1 = W1 + (size_t)l * H_ * FF_;
        const float* w2 = W2 + (size_t)l * FF_ * H_;

        gemm_f16w_kernel<<<gH, blk, 0, stream>>>(x, wq, bq + (size_t)l * H_, qb,
                                                 NTOK, H_, H_, 0);
        gemm_f16w_kernel<<<gH, blk, 0, stream>>>(x, wk, bk + (size_t)l * H_, kb,
                                                 NTOK, H_, H_, 0);
        gemm_f16w_kernel<<<gH, blk, 0, stream>>>(x, wv, bv + (size_t)l * H_, vb,
                                                 NTOK, H_, H_, 0);

        attention_kernel<<<B_ * NH_, blk, 0, stream>>>(qb, kb, vb, amask, ctx);

        gemm_f16w_kernel<<<gH, blk, 0, stream>>>(ctx, wo, bo + (size_t)l * H_, yb,
                                                 NTOK, H_, H_, 0);
        add_ln_kernel<<<NTOK, blk, 0, stream>>>(x, yb, ln1s + (size_t)l * H_,
                                                ln1b + (size_t)l * H_, x);

        gemm_f16w_kernel<<<gF, blk, 0, stream>>>(x, w1, b1 + (size_t)l * FF_, ffb,
                                                 NTOK, FF_, H_, 1);
        gemm_f16w_kernel<<<gH, blk, 0, stream>>>(ffb, w2, b2 + (size_t)l * H_, yb,
                                                 NTOK, H_, FF_, 0);
        add_ln_kernel<<<NTOK, blk, 0, stream>>>(x, yb, ln2s + (size_t)l * H_,
                                                ln2b + (size_t)l * H_, x);
    }

    intent_kernel<<<(B_ * NI_ + 255) / 256, 256, 0, stream>>>(x, Wi, bi, out);
    slot_kernel<<<(B_ * NS_ * LW_) / 256, 256, 0, stream>>>(x, mapping, Ws, bs, out);
}
